// SpatioTemporalBlock_52888227283807
// MI455X (gfx1250) — compile-verified, matrix operand order checked by reading, not run on hardware
//
#include <hip/hip_runtime.h>

typedef __attribute__((ext_vector_type(16))) __bf16 v16bf;
typedef __attribute__((ext_vector_type(8)))  __bf16 v8bf;
typedef __attribute__((ext_vector_type(8)))  float  v8f;

#define N_NODES 20000
#define N_EDGES 640000
#define T_STEPS 64
#define C_IN    64
#define C_S     128
#define C_T     128
#define LN_EPS  1e-5f

union V16 { v16bf v; v8bf h[2]; };

__device__ __forceinline__ __bf16 f2bf(float f) {
    unsigned u = __builtin_bit_cast(unsigned, f);
    unsigned r = u + 0x7FFFu + ((u >> 16) & 1u);          // round-to-nearest-even
    unsigned short hh = (unsigned short)(r >> 16);
    return __builtin_bit_cast(__bf16, hh);
}

__device__ __forceinline__ unsigned pack2bf(float a, float b) {
    unsigned lo = (unsigned)__builtin_bit_cast(unsigned short, f2bf(a));
    unsigned hi = (unsigned)__builtin_bit_cast(unsigned short, f2bf(b));
    return lo | (hi << 16);
}

// K-offset inside a 32-wide chunk for fragment element e, per CDNA5 16-bit A/B layout.
// lanes 0-15:  elems 0..7 -> K 0..7,   elems 8..15 -> K 16..23
// lanes 16-31: elems 0..7 -> K 8..15,  elems 8..15 -> K 24..31
__device__ __forceinline__ int koff(int e, int hi) {
    int base = (e < 8) ? e : (e - 8 + 16);
    return base + (hi ? 8 : 0);
}

// Load a pre-swizzled B fragment: 32 contiguous bytes per lane -> two b128 loads.
__device__ __forceinline__ v16bf load_frag(const __bf16* __restrict__ base, int fragIdx, int lane) {
    const v8bf* p = (const v8bf*)&base[((size_t)fragIdx * 32 + lane) * 16];
    V16 f;
    f.h[0] = p[0];
    f.h[1] = p[1];
    return f.v;
}

// ---------------- degree / dinv ----------------
__global__ void k_deg_init(float* deg) {
    int i = blockIdx.x * blockDim.x + threadIdx.x;
    if (i < N_NODES) deg[i] = 1.0f;                       // self-loop weight
}

__global__ void k_deg_acc(const int* __restrict__ ei, const float* __restrict__ ew,
                          float* __restrict__ deg) {
    int e = blockIdx.x * blockDim.x + threadIdx.x;
    if (e < N_EDGES) atomicAdd(&deg[ei[N_EDGES + e]], ew[e]);
}

__global__ void k_dinv(float* d) {
    int i = blockIdx.x * blockDim.x + threadIdx.x;
    if (i < N_NODES) d[i] = rsqrtf(d[i]);
}

// ---------------- weight prep: fp32 -> bf16, pre-swizzled into WMMA B-fragment order ----
// Swizzled layout: elem = i&15, lane = (i>>4)&31, frag = next bits, colTile = top bits.
// Fragment value = W[k][col] with k = chunkBase + koff(elem, lane>=16), col = colTile*16 + (lane&15).
__global__ void k_prep_weights(const float* __restrict__ Wg, const float* __restrict__ cw,
                               const float* __restrict__ rw,
                               __bf16* __restrict__ Wgb, __bf16* __restrict__ Wkb,
                               __bf16* __restrict__ Rwb) {
    int i = blockIdx.x * blockDim.x + threadIdx.x;
    int e = i & 15, lane = (i >> 4) & 31;
    int hi = lane >> 4, ln = lane & 15;

    if (i < C_IN * C_S) {                                 // W_g: 8 colTiles x 2 K-chunks
        int kk = (i >> 9) & 1, ct = i >> 10;
        int k = kk * 32 + koff(e, hi);
        Wgb[i] = f2bf(Wg[k * C_S + ct * 16 + ln]);
    }
    if (i < C_IN * C_T) {                                 // res_w: 8 colTiles x 2 K-chunks
        int kk = (i >> 9) & 1, ct = i >> 10;
        int k = kk * 32 + koff(e, hi);
        Rwb[i] = f2bf(rw[k * C_T + ct * 16 + ln]);
    }
    if (i < 3 * C_S * C_T) {                              // conv: 8 colTiles x 12 frags (tap,chunk)
        int fid = (i >> 9) % 12, ct = i / (512 * 12);
        int tap = fid >> 2, ch = fid & 3;
        int cs = ch * 32 + koff(e, hi);                   // K dim = input channel
        int oc = ct * 16 + ln;                            // output channel
        Wkb[i] = f2bf(cw[(oc * C_S + cs) * 3 + tap]);
    }
}

// ---------------- aggx = dinv^2 * x  (self-loop term folded into init) ----------------
__global__ __launch_bounds__(256) void k_aggx_init(const float* __restrict__ x,
                                                   const float* __restrict__ dinv,
                                                   float* __restrict__ aggx) {
    size_t i4 = (size_t)blockIdx.x * 256 + threadIdx.x;   // float4 index, 1024 per node
    int n = (int)(i4 >> 10);
    float d = dinv[n], d2 = d * d;
    float4 v = ((const float4*)x)[i4];
    v.x *= d2; v.y *= d2; v.z *= d2; v.w *= d2;
    ((float4*)aggx)[i4] = v;
}

// ---------------- edge scatter: aggx[dst] += norm * x[src] ----------------
__global__ __launch_bounds__(256) void k_edge_scatter(const float* __restrict__ x,
                                                      const int* __restrict__ ei,
                                                      const float* __restrict__ ew,
                                                      const float* __restrict__ dinv,
                                                      float* __restrict__ aggx) {
    const int e = blockIdx.x;
    const int s = ei[e], d = ei[N_EDGES + e];
    const float norm = dinv[s] * ew[e] * dinv[d];
    const float4* xs = (const float4*)(x + (size_t)s * T_STEPS * C_IN);
    float* ad = aggx + (size_t)d * T_STEPS * C_IN;
    const int tid = threadIdx.x;
#pragma unroll
    for (int j = 0; j < 4; ++j) {
        int lin = j * 256 + tid;                          // 1024 float4 per edge payload
        float4 v = xs[lin];
        int b = lin * 4;
        atomicAdd(&ad[b + 0], norm * v.x);
        atomicAdd(&ad[b + 1], norm * v.y);
        atomicAdd(&ad[b + 2], norm * v.z);
        atomicAdd(&ad[b + 3], norm * v.w);
    }
}

// ---------------- s = relu(aggx @ W_g + b_g), bf16 out ----------------
__global__ __launch_bounds__(256) void k_gcn_gemm(const float* __restrict__ aggx,
                                                  const __bf16* __restrict__ Wgb,
                                                  const float* __restrict__ b_g,
                                                  __bf16* __restrict__ s) {
    __shared__ __bf16 As[64 * 72];                        // 64 rows x 64 K, padded stride
    const int tid = threadIdx.x;
    const int rowBase = blockIdx.x * 64;
    const float4* src = (const float4*)(aggx + (size_t)rowBase * C_IN);
#pragma unroll
    for (int j = 0; j < 4; ++j) {
        int lin = j * 256 + tid;                          // 16 float4 per row
        int r = lin >> 4, c4 = (lin & 15) * 4;
        float4 v = src[lin];
        uint2 p;
        p.x = pack2bf(v.x, v.y);
        p.y = pack2bf(v.z, v.w);
        *(uint2*)&As[r * 72 + c4] = p;
    }
    __syncthreads();

    const int wave = tid >> 5, lane = tid & 31;
    const int hi = lane >> 4, ln = lane & 15;
    const int colBase = wave * 16;

    v16bf b0 = load_frag(Wgb, wave * 2 + 0, lane);
    v16bf b1 = load_frag(Wgb, wave * 2 + 1, lane);
    const float bias = b_g[colBase + ln];

#pragma unroll
    for (int rb = 0; rb < 4; ++rb) {
        const __bf16* arow = &As[(rb * 16 + ln) * 72];
        V16 a0, a1;
        a0.h[0] = *(const v8bf*)&arow[hi ? 8 : 0];
        a0.h[1] = *(const v8bf*)&arow[16 + (hi ? 8 : 0)];
        a1.h[0] = *(const v8bf*)&arow[32 + (hi ? 8 : 0)];
        a1.h[1] = *(const v8bf*)&arow[48 + (hi ? 8 : 0)];
        v8f c = {0.f, 0.f, 0.f, 0.f, 0.f, 0.f, 0.f, 0.f};
        c = __builtin_amdgcn_wmma_f32_16x16x32_bf16(false, a0.v, false, b0, (short)0, c, false, false);
        c = __builtin_amdgcn_wmma_f32_16x16x32_bf16(false, a1.v, false, b1, (short)0, c, false, false);
#pragma unroll
        for (int r = 0; r < 8; ++r) {
            int row = rowBase + rb * 16 + r + hi * 8;
            float v = c[r] + bias;
            v = v > 0.f ? v : 0.f;
            s[(size_t)row * C_S + colBase + ln] = f2bf(v);
        }
    }
}

// ---------------- temporal conv1d (K=3, SAME) as 3 shifted WMMA GEMMs ----------------
__global__ __launch_bounds__(256) void k_tconv(const __bf16* __restrict__ s,
                                               const __bf16* __restrict__ Wkb,
                                               const float* __restrict__ conv_b,
                                               float* __restrict__ y) {
    __shared__ __bf16 Ss[66 * 136];                       // rows t=-1..64, padded stride
    const int tid = threadIdx.x;
    const int node = blockIdx.x;
    if (tid < 128) {                                      // zero pad rows
        unsigned* z = (unsigned*)&Ss[(tid < 64 ? 0 : 65) * 136];
        z[tid & 63] = 0u;
    }
    const unsigned* sp = (const unsigned*)(s + (size_t)node * T_STEPS * C_S);
#pragma unroll
    for (int j = 0; j < 16; ++j) {
        int lin = j * 256 + tid;                          // uint idx, 64 per time row
        int t = lin >> 6, cp = lin & 63;
        ((unsigned*)&Ss[(t + 1) * 136])[cp] = sp[lin];
    }
    __syncthreads();

    const int wave = tid >> 5, lane = tid & 31;
    const int hi = lane >> 4, ln = lane & 15;
    const int colBase = wave * 16;

    v16bf bf[12];
#pragma unroll
    for (int f = 0; f < 12; ++f)
        bf[f] = load_frag(Wkb, wave * 12 + f, lane);
    const float bias = conv_b[colBase + ln];

#pragma unroll
    for (int rb = 0; rb < 4; ++rb) {
        v8f c = {0.f, 0.f, 0.f, 0.f, 0.f, 0.f, 0.f, 0.f};
        int trow = rb * 16 + ln;                          // this lane's A-row time
#pragma unroll
        for (int tap = 0; tap < 3; ++tap) {
            const __bf16* arow = &Ss[(trow + tap) * 136]; // time t-1+tap (pad-shifted)
#pragma unroll
            for (int ch = 0; ch < 4; ++ch) {
                V16 a;
                a.h[0] = *(const v8bf*)&arow[ch * 32 + (hi ? 8 : 0)];
                a.h[1] = *(const v8bf*)&arow[ch * 32 + 16 + (hi ? 8 : 0)];
                c = __builtin_amdgcn_wmma_f32_16x16x32_bf16(false, a.v, false, bf[tap * 4 + ch],
                                                            (short)0, c, false, false);
            }
        }
#pragma unroll
        for (int r = 0; r < 8; ++r) {
            int t = rb * 16 + r + hi * 8;
            y[((size_t)node * T_STEPS + t) * C_T + colBase + ln] = c[r] + bias;
        }
    }
}

// ---------------- fused LayerNorm + residual GEMM (in place on d_out) ----------------
__global__ __launch_bounds__(256) void k_ln_res(const float* __restrict__ x,
                                                const __bf16* __restrict__ Rwb,
                                                const float* __restrict__ ln_g,
                                                const float* __restrict__ ln_b,
                                                const float* __restrict__ res_b,
                                                float* __restrict__ out) {
    __shared__ __bf16 Xs[64 * 72];
    __shared__ float red[512];
    __shared__ float muA[64], rsA[64];
    const int tid = threadIdx.x;
    const int rowBase = blockIdx.x * 64;

    const float4* xs = (const float4*)(x + (size_t)rowBase * C_IN);
#pragma unroll
    for (int j = 0; j < 4; ++j) {
        int lin = j * 256 + tid;
        int r = lin >> 4, c4 = (lin & 15) * 4;
        float4 v = xs[lin];
        uint2 p;
        p.x = pack2bf(v.x, v.y);
        p.y = pack2bf(v.z, v.w);
        *(uint2*)&Xs[r * 72 + c4] = p;
    }
    {   // LN stats: 4 threads per row, 32 cols each
        int r = tid >> 2, seg = tid & 3;
        const float* yr = out + ((size_t)(rowBase + r)) * C_T + seg * 32;
        float s0 = 0.f, s1 = 0.f;
#pragma unroll
        for (int i = 0; i < 32; ++i) { float v = yr[i]; s0 += v; s1 += v * v; }
        red[tid] = s0; red[256 + tid] = s1;
    }
    __syncthreads();
    if (tid < 64) {
        int b = tid * 4;
        float s0 = red[b] + red[b + 1] + red[b + 2] + red[b + 3];
        float s1 = red[256 + b] + red[257 + b] + red[258 + b] + red[259 + b];
        float mu = s0 * (1.f / C_T);
        float var = s1 * (1.f / C_T) - mu * mu;
        muA[tid] = mu;
        rsA[tid] = rsqrtf(var + LN_EPS);
    }
    __syncthreads();

    const int wave = tid >> 5, lane = tid & 31;
    const int hi = lane >> 4, ln = lane & 15;
    const int colBase = wave * 16;
    const int col = colBase + ln;

    v16bf b0 = load_frag(Rwb, wave * 2 + 0, lane);
    v16bf b1 = load_frag(Rwb, wave * 2 + 1, lane);
    const float g = ln_g[col], bb = ln_b[col], rbias = res_b[col];

#pragma unroll
    for (int rb = 0; rb < 4; ++rb) {
        const __bf16* arow = &Xs[(rb * 16 + ln) * 72];
        V16 a0, a1;
        a0.h[0] = *(const v8bf*)&arow[hi ? 8 : 0];
        a0.h[1] = *(const v8bf*)&arow[16 + (hi ? 8 : 0)];
        a1.h[0] = *(const v8bf*)&arow[32 + (hi ? 8 : 0)];
        a1.h[1] = *(const v8bf*)&arow[48 + (hi ? 8 : 0)];
        v8f c = {0.f, 0.f, 0.f, 0.f, 0.f, 0.f, 0.f, 0.f};
        c = __builtin_amdgcn_wmma_f32_16x16x32_bf16(false, a0.v, false, b0, (short)0, c, false, false);
        c = __builtin_amdgcn_wmma_f32_16x16x32_bf16(false, a1.v, false, b1, (short)0, c, false, false);
#pragma unroll
        for (int r = 0; r < 8; ++r) {
            int lrow = rb * 16 + r + hi * 8;
            size_t gi = (size_t)(rowBase + lrow) * C_T + col;
            float yv = out[gi];
            float yn = (yv - muA[lrow]) * rsA[lrow] * g + bb;
            out[gi] = yn + c[r] + rbias;
        }
    }
}

extern "C" void kernel_launch(void* const* d_in, const int* in_sizes, int n_in,
                              void* d_out, int out_size, void* d_ws, size_t ws_size,
                              hipStream_t stream) {
    const float* x      = (const float*)d_in[0];
    const int*   ei     = (const int*)d_in[1];
    const float* ew     = (const float*)d_in[2];
    const float* Wg     = (const float*)d_in[3];
    const float* bg     = (const float*)d_in[4];
    const float* convw  = (const float*)d_in[5];
    const float* convb  = (const float*)d_in[6];
    const float* lng    = (const float*)d_in[7];
    const float* lnb    = (const float*)d_in[8];
    const float* resw   = (const float*)d_in[9];
    const float* resb   = (const float*)d_in[10];
    float* out = (float*)d_out;

    char* ws = (char*)d_ws;
    size_t off = 0;
    auto take = [&](size_t bytes) { char* p = ws + off; off = (off + bytes + 255) & ~(size_t)255; return p; };
    float*  dinv  = (float*)take((size_t)N_NODES * 4);
    __bf16* Wgb   = (__bf16*)take((size_t)C_IN * C_S * 2);
    __bf16* Wkb   = (__bf16*)take((size_t)3 * C_S * C_T * 2);
    __bf16* Rwb   = (__bf16*)take((size_t)C_IN * C_T * 2);
    float*  aggx  = (float*)take((size_t)N_NODES * T_STEPS * C_IN * 4);
    __bf16* sbuf  = (__bf16*)take((size_t)N_NODES * T_STEPS * C_S * 2);
    (void)in_sizes; (void)n_in; (void)out_size; (void)ws_size;

    k_deg_init<<<(N_NODES + 255) / 256, 256, 0, stream>>>(dinv);
    k_deg_acc<<<(N_EDGES + 255) / 256, 256, 0, stream>>>(ei, ew, dinv);
    k_dinv<<<(N_NODES + 255) / 256, 256, 0, stream>>>(dinv);
    k_prep_weights<<<(3 * C_S * C_T + 255) / 256, 256, 0, stream>>>(Wg, convw, resw, Wgb, Wkb, Rwb);
    k_aggx_init<<<(N_NODES * T_STEPS * C_IN / 4) / 256, 256, 0, stream>>>(x, dinv, aggx);
    k_edge_scatter<<<N_EDGES, 256, 0, stream>>>(x, ei, ew, dinv, aggx);
    k_gcn_gemm<<<(N_NODES * T_STEPS) / 64, 256, 0, stream>>>(aggx, Wgb, bg, sbuf);
    k_tconv<<<N_NODES, 256, 0, stream>>>(sbuf, Wkb, convb, out);
    k_ln_res<<<(N_NODES * T_STEPS) / 64, 256, 0, stream>>>(x, Rwb, lng, lnb, resb, out);
}
